// DAG_GNN_13194139533783
// MI455X (gfx1250) — matrix operand order repaired and checked by reading, compiled, not run on hardware
//
#include <hip/hip_runtime.h>
#include <hip/hip_bf16.h>

#define Bt   256   // batch
#define NN   64    // nodes
#define KK   8     // neighbors
#define HH   256   // hidden
#define H3   768   // 3*H (GRU gates)
#define LL   3     // layers
#define ZZ   128   // output dim
#define NV   3     // NVAR

typedef __attribute__((ext_vector_type(16))) __bf16 v16bf;
typedef __attribute__((ext_vector_type(8)))  float  v8f;

__device__ __forceinline__ v8f wmma_bf16(v16bf a, v16bf b, v8f c) {
  // v_wmma_f32_16x16x32_bf16: D = A(16x32) * B(32x16) + C(16x16 f32)
  return __builtin_amdgcn_wmma_f32_16x16x32_bf16(false, a, false, b, (short)0, c,
                                                 false, false);
}

// ---- fragment-native layouts ------------------------------------------------
// A tile (16 rows x K cols bf16) is stored in LDS in A-fragment order:
//   element (row b, col h) lives at  (h>>5)*512 + lane*16 + i
//   with lane = b + 16*((h>>3)&1),  i = (h&7) | (((h>>4)&1)<<3)
// so each lane's 16 operands for one WMMA are 32 contiguous bytes.
__device__ __forceinline__ int a_swz(int b, int h) {
  int kt = h >> 5, kk = h & 31;
  int lane = b + (((kk >> 3) & 1) << 4);
  int i = (kk & 7) | (((kk >> 4) & 1) << 3);
  return kt * 512 + lane * 16 + i;
}
__device__ __forceinline__ v16bf load_a_swz(const __bf16* base, int k0, int lane) {
  return *(const v16bf*)(base + (k0 >> 5) * 512 + lane * 16);
}
// B matrices are pre-swizzled in the workspace the same way:
//   S[((kt*nct + ct)*32 + lane)*16 + i] = W[kt*32 + (lane>>4)*8 + (i&7) + (i>>3)*16][ct*16 + (lane&15)]
__device__ __forceinline__ v16bf load_b_swz(const __bf16* W, int nct, int k0,
                                            int col0, int lane) {
  return *(const v16bf*)(W + ((((size_t)(k0 >> 5)) * nct + (col0 >> 4)) * 32 + lane) * 16);
}

// ---- fast activations (hardware v_exp_f32, branchless) ----------------------
__device__ __forceinline__ float fast_sigm(float x) {
  return 1.f / (1.f + __expf(-x));
}
__device__ __forceinline__ float fast_tanh(float x) {
  float xc = fminf(fmaxf(x, -15.f), 15.f);
  float t = __expf(2.f * xc);
  return (t - 1.f) / (t + 1.f);
}
__device__ __forceinline__ float fast_elu(float x) {
  float e = __expf(fminf(x, 0.f)) - 1.f;  // exp on clamped arg: no overflow
  return x > 0.f ? x : e;
}
__device__ __forceinline__ float dot4(float4 a, float4 b) {
  return a.x * b.x + a.y * b.y + a.z * b.z + a.w * b.w;
}

// ---------------------------------------------------------------- prep kernels
__global__ __launch_bounds__(256) void zero_f32(float* __restrict__ p, int n) {
  int i = blockIdx.x * 256 + threadIdx.x;
  if (i < n) p[i] = 0.f;
}

// Swizzle a row-major f32 [K][ncols] weight matrix into B-fragment bf16 order.
__global__ __launch_bounds__(256) void swz_b(const float* __restrict__ in,
                                             __bf16* __restrict__ out,
                                             int K, int ncols) {
  int s = blockIdx.x * 256 + threadIdx.x;
  if (s >= K * ncols) return;
  int i    = s & 15;
  int lane = (s >> 4) & 31;
  int tile = s >> 9;
  int nct  = ncols >> 4;
  int ct   = tile % nct;
  int kt   = tile / nct;
  int k = kt * 32 + ((lane >> 4) << 3) + (i & 7) + ((i >> 3) << 4);
  int n = ct * 16 + (lane & 15);
  out[s] = (__bf16)in[(size_t)k * ncols + n];
}

// out[h] = sum_f W[h*HH + f] * a[f]   (precompute W@a1, W@a2)
__global__ __launch_bounds__(256) void matvec(const float* __restrict__ W,
                                              const float* __restrict__ a,
                                              float* __restrict__ out) {
  int h = threadIdx.x;
  const float4* w4 = (const float4*)(W + (size_t)h * HH);
  const float4* a4 = (const float4*)a;
  float s = 0.f;
#pragma unroll 4
  for (int f = 0; f < HH / 4; ++f) s += dot4(w4[f], a4[f]);
  out[h] = s;
}

// ------------------------------------------------------ fused GAT + GRU step
// Grid: 16 blocks (16 batch rows each), 256 threads (8 wave32s).
__global__ __launch_bounds__(256) void node_step(
    const float* __restrict__ src,     // [B,N,H] gather source
    float* __restrict__ hidden,        // [B,N,H] write row v
    const int* __restrict__ nidx,      // [B,N,K]
    const float* __restrict__ nmask,   // [B,N,K]
    int v,
    const __bf16* __restrict__ Wb,     // GAT W  swizzled [H,H]
    const float* __restrict__ wa1,     // W @ a1 [H]
    const float* __restrict__ wa2,     // W @ a2 [H]
    const __bf16* __restrict__ Wib,    // GRU Wi swizzled [H,3H]
    const __bf16* __restrict__ Whb,    // GRU Wh swizzled [H,3H]
    const float* __restrict__ bi, const float* __restrict__ bh) {
  __shared__ __align__(32) __bf16 s_hs[16 * HH];   // swizzled A tiles
  __shared__ __align__(32) __bf16 s_agg[16 * HH];
  __shared__ __align__(32) __bf16 s_hvb[16 * HH];
  __shared__ float  s_hvf[16 * HH];    // hv f32 for gate combine
  __shared__ float  s_am[16 * KK];     // alpha * mask
  __shared__ float  s_en[16 * KK];     // neighbor logits
  __shared__ int    s_id[16 * KK];
  __shared__ float  s_mk[16 * KK];
  __shared__ float  s_es[16];          // hs . wa1

  const int tid = threadIdx.x;
  const int bg  = blockIdx.x * 16;  // batch base

  __builtin_prefetch(Wb, 0, 1);
  __builtin_prefetch(Wib, 0, 1);
  __builtin_prefetch(Whb, 0, 1);

  // ---- phase 1: indices/masks + hs tile (vectorized, swizzled store)
  if (tid < 16 * KK) {
    int b = tid >> 3, k = tid & 7;
    s_id[tid] = nidx[((size_t)(bg + b) * NN + v) * KK + k];
    s_mk[tid] = nmask[((size_t)(bg + b) * NN + v) * KK + k];
  }
  {
    // 4096 elems as 1024 float4 chunks over 256 threads
    int b0 = tid >> 6, h4 = (tid & 63) << 2;
    for (int b = b0; b < 16; b += 4) {
      float4 val = *(const float4*)(src + ((size_t)(bg + b) * NN + v) * HH + h4);
      s_hs[a_swz(b, h4 + 0)] = (__bf16)val.x;
      s_hs[a_swz(b, h4 + 1)] = (__bf16)val.y;
      s_hs[a_swz(b, h4 + 2)] = (__bf16)val.z;
      s_hs[a_swz(b, h4 + 3)] = (__bf16)val.w;
    }
  }
  __syncthreads();

  // attention dots: e_nbr = mask * (src_row . wa2), e_self = hs . wa1
  if (tid < 128) {
    int b = tid >> 3;
    const float4* row = (const float4*)(src + ((size_t)(bg + b) * NN + s_id[tid]) * HH);
    const float4* w4  = (const float4*)wa2;
    float s = 0.f;
#pragma unroll 4
    for (int h = 0; h < HH / 4; ++h) s += dot4(row[h], w4[h]);
    s_en[tid] = s * s_mk[tid];
  } else if (tid < 144) {
    int b = tid - 128;
    const float4* row = (const float4*)(src + ((size_t)(bg + b) * NN + v) * HH);
    const float4* w4  = (const float4*)wa1;
    float s = 0.f;
#pragma unroll 4
    for (int h = 0; h < HH / 4; ++h) s += dot4(row[h], w4[h]);
    s_es[b] = s;
  }
  __syncthreads();

  // leaky-relu -> masked softmax -> alpha*mask (degree-0 rows zeroed)
  if (tid < 16) {
    int b = tid;
    float deg = 0.f, mx = -1e30f, e[KK];
#pragma unroll
    for (int k = 0; k < KK; ++k) {
      float m = s_mk[b * KK + k];
      deg += m;
      float val = s_es[b] + s_en[b * KK + k];
      val = fmaxf(val, 0.2f * val);      // leaky_relu(0.2), branchless
      val = (m > 0.f) ? val : -1e9f;
      e[k] = val;
      mx = fmaxf(mx, val);
    }
    float den = 0.f;
#pragma unroll
    for (int k = 0; k < KK; ++k) { e[k] = __expf(e[k] - mx); den += e[k]; }
    float keep = deg > 0.f ? 1.f : 0.f;
    float rden = keep / den;
#pragma unroll
    for (int k = 0; k < KK; ++k)
      s_am[b * KK + k] = e[k] * rden * s_mk[b * KK + k];
  }
  __syncthreads();

  // agg[b,h] = sum_k alpha*mask * src[idx_k, h]  (float4 gather, swizzled store)
  {
    int b0 = tid >> 6, h4 = (tid & 63) << 2;
    for (int b = b0; b < 16; b += 4) {
      float4 acc = make_float4(0.f, 0.f, 0.f, 0.f);
#pragma unroll
      for (int k = 0; k < KK; ++k) {
        float a = s_am[b * KK + k];
        float4 val = *(const float4*)(src + ((size_t)(bg + b) * NN + s_id[b * KK + k]) * HH + h4);
        acc.x += a * val.x; acc.y += a * val.y;
        acc.z += a * val.z; acc.w += a * val.w;
      }
      s_agg[a_swz(b, h4 + 0)] = (__bf16)acc.x;
      s_agg[a_swz(b, h4 + 1)] = (__bf16)acc.y;
      s_agg[a_swz(b, h4 + 2)] = (__bf16)acc.z;
      s_agg[a_swz(b, h4 + 3)] = (__bf16)acc.w;
    }
  }
  __syncthreads();

  const int wave = tid >> 5, lane = tid & 31, half = lane >> 4;

  // ---- phase 2a: hv = elu(agg @ W), 16 col-tiles over 8 waves
  for (int t = wave; t < 16; t += 8) {
    v8f c = {};
#pragma unroll
    for (int k0 = 0; k0 < HH; k0 += 32) {
      v16bf a  = load_a_swz(s_agg, k0, lane);
      v16bf bm = load_b_swz(Wb, HH / 16, k0, t * 16, lane);
      c = wmma_bf16(a, bm, c);
    }
    const int col = t * 16 + (lane & 15);
#pragma unroll
    for (int r = 0; r < 8; ++r) {
      float val = fast_elu(c[r]);
      int M = r + 8 * half;
      s_hvf[M * HH + col]  = val;
      s_hvb[a_swz(M, col)] = (__bf16)val;
    }
  }
  __syncthreads();

  // ---- phase 2b: fused GRU — 6 accumulators/wave (r,z,n gates x Wi,Wh)
  for (int t = wave; t < 16; t += 8) {
    v8f cir = {}, ciz = {}, cin = {}, chr_ = {}, chz = {}, chn = {};
#pragma unroll
    for (int k0 = 0; k0 < HH; k0 += 32) {
      v16bf ax = load_a_swz(s_hs,  k0, lane);
      v16bf ah = load_a_swz(s_hvb, k0, lane);
      v16bf br = load_b_swz(Wib, H3 / 16, k0, t * 16,       lane);
      v16bf bz = load_b_swz(Wib, H3 / 16, k0, 256 + t * 16, lane);
      v16bf bn = load_b_swz(Wib, H3 / 16, k0, 512 + t * 16, lane);
      cir = wmma_bf16(ax, br, cir);
      ciz = wmma_bf16(ax, bz, ciz);
      cin = wmma_bf16(ax, bn, cin);
      br = load_b_swz(Whb, H3 / 16, k0, t * 16,       lane);
      bz = load_b_swz(Whb, H3 / 16, k0, 256 + t * 16, lane);
      bn = load_b_swz(Whb, H3 / 16, k0, 512 + t * 16, lane);
      chr_ = wmma_bf16(ah, br, chr_);
      chz  = wmma_bf16(ah, bz, chz);
      chn  = wmma_bf16(ah, bn, chn);
    }
    const int col = t * 16 + (lane & 15);
    const float bir = bi[col], biz = bi[256 + col], bin = bi[512 + col];
    const float bhr = bh[col], bhz = bh[256 + col], bhn = bh[512 + col];
#pragma unroll
    for (int r = 0; r < 8; ++r) {
      int M = r + 8 * half;
      float hv = s_hvf[M * HH + col];
      float rg = fast_sigm(cir[r] + bir + chr_[r] + bhr);
      float zg = fast_sigm(ciz[r] + biz + chz[r] + bhz);
      float ng = fast_tanh(cin[r] + bin + rg * (chn[r] + bhn));
      hidden[((size_t)(bg + M) * NN + v) * HH + col] = (1.f - zg) * ng + zg * hv;
    }
  }
}

// -------------------------------------------------- variable readout per layer
__global__ __launch_bounds__(256) void readout(const float* __restrict__ hidden,
                                               const int* __restrict__ var_pos,
                                               float* __restrict__ vo_l) {
  int row = blockIdx.x;  // b*NV + j, 768 rows
  int b = row / NV, j = row % NV;
  int p = var_pos[b * NV + j];
  vo_l[(size_t)row * HH + threadIdx.x] =
      hidden[((size_t)b * NN + p) * HH + threadIdx.x];
}

// ------------------------------------------- 3-step GRU over the layer axis
// Grid: 48 blocks (16 of 768 rows each), 256 threads.
__global__ __launch_bounds__(256) void gru_var(
    const float* __restrict__ vo,      // [L][768][H]
    const __bf16* __restrict__ Wib, const __bf16* __restrict__ Whb,  // swizzled [H,3H]
    const float* __restrict__ bi, const float* __restrict__ bh,
    float* __restrict__ hvar) {        // [768][H]
  __shared__ __align__(32) __bf16 s_x[16 * HH];
  __shared__ __align__(32) __bf16 s_hb[16 * HH];
  __shared__ float  s_hf[16 * HH];
  const int tid = threadIdx.x;
  const int rg0 = blockIdx.x * 16;

  for (int e = tid; e < 16 * HH; e += 256) s_hf[e] = 0.f;
  __syncthreads();

  for (int t = 0; t < LL; ++t) {
    for (int e = tid; e < 16 * HH; e += 256) {
      int rr = e >> 8, h = e & 255;
      int sa = a_swz(rr, h);
      s_x[sa]  = (__bf16)vo[((size_t)t * (Bt * NV) + rg0 + rr) * HH + h];
      s_hb[sa] = (__bf16)s_hf[e];
    }
    __syncthreads();

    const int wave = tid >> 5, lane = tid & 31, half = lane >> 4;
    for (int t2 = wave; t2 < 16; t2 += 8) {
      v8f cir = {}, ciz = {}, cin = {}, chr_ = {}, chz = {}, chn = {};
#pragma unroll
      for (int k0 = 0; k0 < HH; k0 += 32) {
        v16bf ax = load_a_swz(s_x,  k0, lane);
        v16bf ah = load_a_swz(s_hb, k0, lane);
        v16bf br = load_b_swz(Wib, H3 / 16, k0, t2 * 16,       lane);
        v16bf bz = load_b_swz(Wib, H3 / 16, k0, 256 + t2 * 16, lane);
        v16bf bn = load_b_swz(Wib, H3 / 16, k0, 512 + t2 * 16, lane);
        cir = wmma_bf16(ax, br, cir);
        ciz = wmma_bf16(ax, bz, ciz);
        cin = wmma_bf16(ax, bn, cin);
        br = load_b_swz(Whb, H3 / 16, k0, t2 * 16,       lane);
        bz = load_b_swz(Whb, H3 / 16, k0, 256 + t2 * 16, lane);
        bn = load_b_swz(Whb, H3 / 16, k0, 512 + t2 * 16, lane);
        chr_ = wmma_bf16(ah, br, chr_);
        chz  = wmma_bf16(ah, bz, chz);
        chn  = wmma_bf16(ah, bn, chn);
      }
      const int col = t2 * 16 + (lane & 15);
      const float bir = bi[col], biz = bi[256 + col], bin = bi[512 + col];
      const float bhr = bh[col], bhz = bh[256 + col], bhn = bh[512 + col];
#pragma unroll
      for (int r = 0; r < 8; ++r) {
        int M = r + 8 * half;
        float h0 = s_hf[M * HH + col];
        float rg = fast_sigm(cir[r] + bir + chr_[r] + bhr);
        float zg = fast_sigm(ciz[r] + biz + chz[r] + bhz);
        float ng = fast_tanh(cin[r] + bin + rg * (chn[r] + bhn));
        s_hf[M * HH + col] = (1.f - zg) * ng + zg * h0;
      }
    }
    __syncthreads();
  }
  for (int e = tid; e < 16 * HH; e += 256)
    hvar[((size_t)rg0 + (e >> 8)) * HH + (e & 255)] = s_hf[e];
}

// ------------------------------------------------------------------- final MLP
// out[256,128] = reshape(hvar,[256,768]) @ mlp_W + b. Grid: 16 blocks.
__global__ __launch_bounds__(256) void mlp_out(
    const float* __restrict__ hvar,   // [768][H], row = b*NV + j
    const __bf16* __restrict__ Wb,    // swizzled [768][128]
    const float* __restrict__ bias, float* __restrict__ out) {
  __shared__ __align__(32) __bf16 s_a[16 * H3];  // 24 k-tiles * 512
  const int tid = threadIdx.x;
  const int m0  = blockIdx.x * 16;  // batch rows

  for (int e = tid; e < 16 * H3; e += 256) {
    int b = e / H3, c = e % H3;
    s_a[a_swz(b, c)] =
        (__bf16)hvar[((size_t)(m0 + b) * NV + c / HH) * HH + (c % HH)];
  }
  __syncthreads();

  const int wave = tid >> 5, lane = tid & 31, half = lane >> 4;
  v8f c = {};
#pragma unroll 4
  for (int k0 = 0; k0 < H3; k0 += 32) {
    v16bf a  = load_a_swz(s_a, k0, lane);
    v16bf bm = load_b_swz(Wb, ZZ / 16, k0, wave * 16, lane);
    c = wmma_bf16(a, bm, c);
  }
  const int col = wave * 16 + (lane & 15);
  const float bb = bias[col];
#pragma unroll
  for (int r = 0; r < 8; ++r) {
    int M = r + 8 * half;
    out[(size_t)(m0 + M) * ZZ + col] = c[r] + bb;
  }
}

// ================================================================== launcher
extern "C" void kernel_launch(void* const* d_in, const int* in_sizes, int n_in,
                              void* d_out, int out_size, void* d_ws, size_t ws_size,
                              hipStream_t stream) {
  const float* x          = (const float*)d_in[0];
  const int*   pred_idx   = (const int*)d_in[1];
  const float* pred_mask  = (const float*)d_in[2];
  const int*   child_idx  = (const int*)d_in[3];
  const float* child_mask = (const float*)d_in[4];
  const int*   var_pos    = (const int*)d_in[5];
  const float* gatf_W  = (const float*)d_in[6];
  const float* gatf_a1 = (const float*)d_in[7];
  const float* gatf_a2 = (const float*)d_in[8];
  const float* gatr_W  = (const float*)d_in[9];
  const float* gatr_a1 = (const float*)d_in[10];
  const float* gatr_a2 = (const float*)d_in[11];
  const float* gruf_Wi = (const float*)d_in[12];
  const float* gruf_Wh = (const float*)d_in[13];
  const float* gruf_bi = (const float*)d_in[14];
  const float* gruf_bh = (const float*)d_in[15];
  const float* grub_Wi = (const float*)d_in[16];
  const float* grub_Wh = (const float*)d_in[17];
  const float* grub_bi = (const float*)d_in[18];
  const float* grub_bh = (const float*)d_in[19];
  const float* gruv_Wi = (const float*)d_in[20];
  const float* gruv_Wh = (const float*)d_in[21];
  const float* gruv_bi = (const float*)d_in[22];
  const float* gruv_bh = (const float*)d_in[23];
  const float* mlp_W   = (const float*)d_in[24];
  const float* mlp_b   = (const float*)d_in[25];
  float* out = (float*)d_out;

  // ---- carve workspace (256B aligned)
  char* wp = (char*)d_ws;
  auto alloc = [&](size_t bytes) -> void* {
    void* p = wp;
    wp += (bytes + 255) & ~(size_t)255;
    return p;
  };
  __bf16* gatfWb  = (__bf16*)alloc((size_t)LL * HH * HH * 2);
  __bf16* gatrWb  = (__bf16*)alloc((size_t)LL * HH * HH * 2);
  __bf16* grufWib = (__bf16*)alloc((size_t)LL * HH * H3 * 2);
  __bf16* grufWhb = (__bf16*)alloc((size_t)LL * HH * H3 * 2);
  __bf16* grubWib = (__bf16*)alloc((size_t)LL * HH * H3 * 2);
  __bf16* grubWhb = (__bf16*)alloc((size_t)LL * HH * H3 * 2);
  __bf16* gruvWib = (__bf16*)alloc((size_t)HH * H3 * 2);
  __bf16* gruvWhb = (__bf16*)alloc((size_t)HH * H3 * 2);
  __bf16* mlpWb   = (__bf16*)alloc((size_t)H3 * ZZ * 2);
  float* wa1f = (float*)alloc((size_t)LL * HH * 4);
  float* wa2f = (float*)alloc((size_t)LL * HH * 4);
  float* wa1r = (float*)alloc((size_t)LL * HH * 4);
  float* wa2r = (float*)alloc((size_t)LL * HH * 4);
  float* vo     = (float*)alloc((size_t)LL * Bt * NV * HH * 4);
  float* hvar   = (float*)alloc((size_t)Bt * NV * HH * 4);
  float* hidden = (float*)alloc((size_t)Bt * NN * HH * 4);

  // Swizzle one [K][ncols] matrix (or L stacked ones) into B-fragment order.
  auto swz = [&](const float* src, __bf16* dst, int K, int ncols, int count) {
    int n = K * ncols;
    for (int l = 0; l < count; ++l)
      swz_b<<<(n + 255) / 256, 256, 0, stream>>>(src + (size_t)l * n,
                                                 dst + (size_t)l * n, K, ncols);
  };

  // ---- prep: zero hidden, swizzled bf16 weights, w_a1/w_a2 projections
  {
    int n = Bt * NN * HH;
    zero_f32<<<(n + 255) / 256, 256, 0, stream>>>(hidden, n);
  }
  swz(gatf_W,  gatfWb,  HH, HH, LL);
  swz(gatr_W,  gatrWb,  HH, HH, LL);
  swz(gruf_Wi, grufWib, HH, H3, LL);
  swz(gruf_Wh, grufWhb, HH, H3, LL);
  swz(grub_Wi, grubWib, HH, H3, LL);
  swz(grub_Wh, grubWhb, HH, H3, LL);
  swz(gruv_Wi, gruvWib, HH, H3, 1);
  swz(gruv_Wh, gruvWhb, HH, H3, 1);
  swz(mlp_W,   mlpWb,   H3, ZZ, 1);
  for (int l = 0; l < LL; ++l) {
    matvec<<<1, 256, 0, stream>>>(gatf_W + (size_t)l * HH * HH, gatf_a1 + l * HH, wa1f + l * HH);
    matvec<<<1, 256, 0, stream>>>(gatf_W + (size_t)l * HH * HH, gatf_a2 + l * HH, wa2f + l * HH);
    matvec<<<1, 256, 0, stream>>>(gatr_W + (size_t)l * HH * HH, gatr_a1 + l * HH, wa1r + l * HH);
    matvec<<<1, 256, 0, stream>>>(gatr_W + (size_t)l * HH * HH, gatr_a2 + l * HH, wa2r + l * HH);
  }

  // ---- L layers: forward sweep, readout, (reverse sweep)
  for (int l = 0; l < LL; ++l) {
    const float* src = (l == 0) ? x : hidden;
    for (int v = 0; v < NN; ++v) {
      node_step<<<16, 256, 0, stream>>>(
          src, hidden, pred_idx, pred_mask, v,
          gatfWb + (size_t)l * HH * HH, wa1f + l * HH, wa2f + l * HH,
          grufWib + (size_t)l * HH * H3, grufWhb + (size_t)l * HH * H3,
          gruf_bi + l * H3, gruf_bh + l * H3);
    }
    readout<<<Bt * NV, 256, 0, stream>>>(hidden, var_pos, vo + (size_t)l * Bt * NV * HH);
    if (l != LL - 1) {
      for (int v = NN - 1; v >= 0; --v) {
        node_step<<<16, 256, 0, stream>>>(
            hidden, hidden, child_idx, child_mask, v,
            gatrWb + (size_t)l * HH * HH, wa1r + l * HH, wa2r + l * HH,
            grubWib + (size_t)l * HH * H3, grubWhb + (size_t)l * HH * H3,
            grub_bi + l * H3, grub_bh + l * H3);
      }
    }
  }

  // ---- variable GRU over layer axis + final MLP
  gru_var<<<(Bt * NV) / 16, 256, 0, stream>>>(vo, gruvWib, gruvWhb, gruv_bi, gruv_bh, hvar);
  mlp_out<<<Bt / 16, 256, 0, stream>>>(hvar, mlpWb, mlp_b, out);

  (void)in_sizes; (void)n_in; (void)out_size; (void)ws_size;
}
